// DinoVisionTransformerSparseMoEFC2_LT_25701084299304
// MI455X (gfx1250) — matrix operand order repaired and checked by reading, not compiled, not run on hardware
//
#include <hip/hip_runtime.h>
#include <hip/hip_bf16.h>
#include <stdint.h>

// Problem constants (from reference setup_inputs): B=8,N=1024 -> T=8192
#define T_TOK 8192
#define HDIM  4096
#define DDIM  1024
#define EDIM  8
#define RDIM  64
#define LRDIM (EDIM * RDIM)   // 512

typedef __attribute__((ext_vector_type(16))) __bf16 v16bf;
typedef __attribute__((ext_vector_type(8)))  float  v8f;

// ---- CDNA5 async global->LDS path (probe at compile time, safe fallback) --
#ifndef USE_ASYNC_LDS
# if defined(__has_builtin)
#  if __has_builtin(__builtin_amdgcn_global_load_async_to_lds_b128) && \
      __has_builtin(__builtin_amdgcn_s_wait_asynccnt)
#   define USE_ASYNC_LDS 1
#  else
#   define USE_ASYNC_LDS 0
#  endif
# else
#  define USE_ASYNC_LDS 0
# endif
#endif

#if USE_ASYNC_LDS
// builtin signature (from hipcc diagnostic): param0 = v4i in AS(1), b128 payload
typedef __attribute__((__vector_size__(4 * sizeof(int)))) int v4i;
typedef __attribute__((address_space(1))) v4i as1_v4i;
typedef __attribute__((address_space(3))) v4i as3_v4i;
#endif

__device__ __forceinline__ void stage_issue(const unsigned short* __restrict__ src,
                                            unsigned short* __restrict__ dst) {
#if USE_ASYNC_LDS
    __builtin_amdgcn_global_load_async_to_lds_b128((as1_v4i*)src, (as3_v4i*)dst, 0, 0);
#else
    *(uint4*)dst = *(const uint4*)src;
#endif
}

template <int N>
__device__ __forceinline__ void stage_wait_cnt() {
#if USE_ASYNC_LDS
    __builtin_amdgcn_s_wait_asynccnt(N);   // immediate; in-order async completion
#endif
}

__device__ __forceinline__ unsigned short f32_to_bf16(float f) {
    unsigned int u = __float_as_uint(f);
    unsigned int r = u + 0x7FFFu + ((u >> 16) & 1u);   // round to nearest even
    if ((u & 0x7F800000u) == 0x7F800000u) r = u;       // keep Inf/NaN payload
    return (unsigned short)(r >> 16);
}

// ---------------- router: logits -> top-2 -> renormalized combine*scale ----
__global__ void router_kernel(const float* __restrict__ x,
                              const float* __restrict__ Wg,
                              const float* __restrict__ scale,
                              float* __restrict__ cwscale) {
    int wid  = (blockIdx.x * blockDim.x + threadIdx.x) >> 5;  // one wave32 / token
    int lane = threadIdx.x & 31;
    if (wid >= T_TOK) return;
    const float* xr = x + (size_t)wid * HDIM;
    float acc[EDIM];
#pragma unroll
    for (int e = 0; e < EDIM; ++e) acc[e] = 0.f;
    for (int h = lane; h < HDIM; h += 32) {
        float xv = xr[h];
        const float4* wr = (const float4*)(Wg + (size_t)h * EDIM);
        float4 a = wr[0], b = wr[1];
        acc[0] += xv * a.x; acc[1] += xv * a.y; acc[2] += xv * a.z; acc[3] += xv * a.w;
        acc[4] += xv * b.x; acc[5] += xv * b.y; acc[6] += xv * b.z; acc[7] += xv * b.w;
    }
#pragma unroll
    for (int off = 16; off > 0; off >>= 1) {
#pragma unroll
        for (int e = 0; e < EDIM; ++e) acc[e] += __shfl_xor(acc[e], off, 32);
    }
    if (lane == 0) {
        int i0 = 0;
#pragma unroll
        for (int e = 1; e < EDIM; ++e) if (acc[e] > acc[i0]) i0 = e;
        int i1 = (i0 == 0) ? 1 : 0;
#pragma unroll
        for (int e = 0; e < EDIM; ++e) {
            if (e == i0 || e == i1) continue;
            if (acc[e] > acc[i1]) i1 = e;
        }
        float ed = __expf(acc[i1] - acc[i0]);   // softmax top-2 renormalized
        float w0 = 1.f / (1.f + ed);
        float w1 = ed / (1.f + ed);
        float* o = cwscale + (size_t)wid * EDIM;
#pragma unroll
        for (int e = 0; e < EDIM; ++e) {
            float cw = (e == i0) ? w0 : (e == i1) ? w1 : 0.f;
            o[e] = cw * scale[e];
        }
    }
}

// ---------------- x conversion: 8 floats -> 8 bf16 per iteration ----------
__global__ void cvt_f32_bf16_v8(const float* __restrict__ in,
                                unsigned short* __restrict__ out, long n8) {
    long i = blockIdx.x * (long)blockDim.x + threadIdx.x;
    long stride = (long)gridDim.x * blockDim.x;
    for (; i < n8; i += stride) {
        const float4* p = (const float4*)(in + i * 8);
        float4 a = p[0], b = p[1];
        union { unsigned short u[8]; uint4 q; } o;
        o.u[0] = f32_to_bf16(a.x); o.u[1] = f32_to_bf16(a.y);
        o.u[2] = f32_to_bf16(a.z); o.u[3] = f32_to_bf16(a.w);
        o.u[4] = f32_to_bf16(b.x); o.u[5] = f32_to_bf16(b.y);
        o.u[6] = f32_to_bf16(b.z); o.u[7] = f32_to_bf16(b.w);
        *(uint4*)(out + i * 8) = o.q;
    }
}

// ---------------- B-side packing into WMMA fragment order -----------------
// Packed layout: Bp[(kb * N + n)*32 + kk] = B[kb*32 + kk, n]  (bf16)
// => one K-block x 64 columns is a contiguous 4 KB slab; a lane's fragment
//    (col n, K = k0 + hi*16 + 0..15) is 32 contiguous bytes.

__global__ void pack_W2(const float* __restrict__ W2, unsigned short* __restrict__ Wp) {
    long n = (long)HDIM * DDIM;
    long i = blockIdx.x * (long)blockDim.x + threadIdx.x;
    long stride = (long)gridDim.x * blockDim.x;
    for (; i < n; i += stride) {
        long kb  = i / ((long)DDIM * 32);
        long rem = i % ((long)DDIM * 32);
        int  col = (int)(rem >> 5);
        int  kk  = (int)(rem & 31);
        long k   = kb * 32 + kk;
        Wp[i] = f32_to_bf16(W2[k * DDIM + col]);
    }
}

__global__ void pack_A(const float* __restrict__ A, unsigned short* __restrict__ Ap) {
    long n = (long)HDIM * LRDIM;
    long i = blockIdx.x * (long)blockDim.x + threadIdx.x;
    long stride = (long)gridDim.x * blockDim.x;
    for (; i < n; i += stride) {
        long kb  = i / ((long)LRDIM * 32);
        long rem = i % ((long)LRDIM * 32);
        int  col = (int)(rem >> 5);            // e*64 + r
        int  kk  = (int)(rem & 31);
        long h   = kb * 32 + kk;
        int  e   = col >> 6, r = col & 63;
        Ap[i] = f32_to_bf16(A[((size_t)e * HDIM + h) * RDIM + r]);
    }
}

__global__ void pack_Bm(const float* __restrict__ Bm, unsigned short* __restrict__ Bp) {
    long n = (long)LRDIM * DDIM;
    long i = blockIdx.x * (long)blockDim.x + threadIdx.x;
    long stride = (long)gridDim.x * blockDim.x;
    for (; i < n; i += stride) {
        long kb  = i / ((long)DDIM * 32);
        long rem = i % ((long)DDIM * 32);
        int  col = (int)(rem >> 5);
        int  kk  = (int)(rem & 31);
        long k   = kb * 32 + kk;               // e*64 + r
        int  e   = (int)(k >> 6), r = (int)(k & 63);
        Bp[i] = f32_to_bf16(Bm[((size_t)e * RDIM + r) * DDIM + col]);
    }
}

// ---------------- WMMA fragment loaders -----------------------------------
// A (16x32 bf16, row-major source): lane row = m0 + (lane&15);
// lanes<16 hold K {k0..k0+15}, lanes>=16 hold {k0+16..k0+31}.
__device__ __forceinline__ v16bf load_afrag(const unsigned short* __restrict__ base,
                                            int ldk, int row, int k0, int hi) {
    union { uint4 q[2]; v16bf v; } f;
    const unsigned short* p = base + (size_t)row * ldk + k0 + (hi ? 16 : 0);
    f.q[0] = *(const uint4*)(p);
    f.q[1] = *(const uint4*)(p + 8);
    return f.v;
}

// B (32x16 bf16) fragment from a staged LDS slab: lc = local column 0..63.
__device__ __forceinline__ v16bf lds_bfrag(const unsigned short* lb, int lc, int hi) {
    union { uint4 q[2]; v16bf v; } f;
    const unsigned short* p = lb + lc * 32 + hi * 16;
    f.q[0] = *(const uint4*)(p);
    f.q[1] = *(const uint4*)(p + 8);
    return f.v;
}

// One K-step: A frag from global, 4 B frags from the staged LDS slab.
__device__ __forceinline__ void compute_step(const unsigned short* __restrict__ Amat, int lda,
                                             int m_row, int hi, int nl, int kb,
                                             const unsigned short* lbc, v8f* c) {
    v16bf a = load_afrag(Amat, lda, m_row, kb * 32, hi);
#pragma unroll
    for (int j = 0; j < 4; ++j) {
        v16bf b = lds_bfrag(lbc, j * 16 + nl, hi);
        c[j] = __builtin_amdgcn_wmma_f32_16x16x32_bf16(false, a, false, b,
                                                       (short)0, c[j], false, false);
    }
}

// ---------------- shared GEMM phase: 4-slab async pipeline ----------------
// Workgroup: 8 waves = 8 M-tiles (128 rows) x one 64-wide N-quad.
// Copies run 2 K-steps ahead of consumption (steady wait: asynccnt <= 2).
// 4 buffers: write target (kb+2)%4 never aliases the oldest live reader
// (kb-1)%4 because waves drift < 1 barrier interval and 3 % 4 != 0.
__device__ __forceinline__ void gemm_phase(const unsigned short* __restrict__ Amat, int lda,
                                           const unsigned short* __restrict__ Bpk, int ldb,
                                           int n0, int KB, int m_row, int hi, int nl,
                                           int tid, unsigned short* lb, v8f* c) {
    stage_issue(Bpk + ((size_t)0 * ldb + n0) * 32 + tid * 8, lb + 0 * 2048 + tid * 8);
    stage_issue(Bpk + ((size_t)1 * ldb + n0) * 32 + tid * 8, lb + 1 * 2048 + tid * 8);
    int kb = 0;
    for (; kb < KB - 2; ++kb) {                       // steady state: no branches inside
        stage_issue(Bpk + ((size_t)(kb + 2) * ldb + n0) * 32 + tid * 8,
                    lb + ((kb + 2) & 3) * 2048 + tid * 8);
        stage_wait_cnt<2>();                          // slab kb has landed (in-order)
        __syncthreads();
        compute_step(Amat, lda, m_row, hi, nl, kb, lb + (kb & 3) * 2048, c);
    }
    stage_wait_cnt<1>();                              // peeled tail: kb = KB-2
    __syncthreads();
    compute_step(Amat, lda, m_row, hi, nl, kb, lb + (kb & 3) * 2048, c);
    ++kb;                                             // kb = KB-1
    stage_wait_cnt<0>();
    __syncthreads();
    compute_step(Amat, lda, m_row, hi, nl, kb, lb + (kb & 3) * 2048, c);
    __syncthreads();                                  // phase-boundary safety
}

// ---------------- GEMM 1: h' = (x @ A_all) * cwscale, stored bf16 ---------
__global__ void lora_h_kernel(const unsigned short* __restrict__ xb,
                              const unsigned short* __restrict__ Ap,
                              const float* __restrict__ cwscale,
                              unsigned short* __restrict__ hprime) {
    __shared__ unsigned short lb[4 * 2048];            // 4 x 4 KB B slabs
    int tid = threadIdx.x;
    int wv = tid >> 5, lane = tid & 31, hi = lane >> 4, nl = lane & 15;
    const int NQ = LRDIM / 64;                         // 8 quads (one per expert)
    int bq = blockIdx.x % NQ;
    int bm = blockIdx.x / NQ;
    int n0 = bq * 64;
    int m0 = bm * 128 + wv * 16;
    int e  = bq;
    v8f c[4];
#pragma unroll
    for (int j = 0; j < 4; ++j) c[j] = (v8f){0.f,0.f,0.f,0.f,0.f,0.f,0.f,0.f};
    gemm_phase(xb, HDIM, Ap, LRDIM, n0, HDIM / 32, m0 + nl, hi, nl, tid, lb, c);
#pragma unroll
    for (int i = 0; i < 8; ++i) {
        int row = m0 + hi * 8 + i;                     // C layout: M = i + 8*hi
        float w = cwscale[(size_t)row * EDIM + e];
        unsigned short* hr = hprime + (size_t)row * LRDIM + n0 + nl;
#pragma unroll
        for (int j = 0; j < 4; ++j) hr[j * 16] = f32_to_bf16(c[j][i] * w);
    }
}

// ---------------- GEMM 2: out = x@W2 + h'@B_all + b2 ----------------------
__global__ void out_kernel(const unsigned short* __restrict__ xb,
                           const unsigned short* __restrict__ Wp,
                           const unsigned short* __restrict__ hprime,
                           const unsigned short* __restrict__ Bp,
                           const float* __restrict__ b2,
                           float* __restrict__ out) {
    __shared__ unsigned short lb[4 * 2048];
    int tid = threadIdx.x;
    int wv = tid >> 5, lane = tid & 31, hi = lane >> 4, nl = lane & 15;
    const int NQ = DDIM / 64;                          // 16 quads
    int bq = blockIdx.x % NQ;
    int bm = blockIdx.x / NQ;
    int n0 = bq * 64;
    int m0 = bm * 128 + wv * 16;
    v8f c[4];
#pragma unroll
    for (int j = 0; j < 4; ++j) c[j] = (v8f){0.f,0.f,0.f,0.f,0.f,0.f,0.f,0.f};
    gemm_phase(xb,     HDIM,  Wp, DDIM, n0, HDIM  / 32, m0 + nl, hi, nl, tid, lb, c); // base
    gemm_phase(hprime, LRDIM, Bp, DDIM, n0, LRDIM / 32, m0 + nl, hi, nl, tid, lb, c); // LoRA
#pragma unroll
    for (int i = 0; i < 8; ++i) {
        int row = m0 + hi * 8 + i;
        float* orow = out + (size_t)row * DDIM + n0 + nl;
#pragma unroll
        for (int j = 0; j < 4; ++j) orow[j * 16] = c[j][i] + b2[n0 + j * 16 + nl];
    }
}

// ---------------- host launcher -------------------------------------------
extern "C" void kernel_launch(void* const* d_in, const int* in_sizes, int n_in,
                              void* d_out, int out_size, void* d_ws, size_t ws_size,
                              hipStream_t stream) {
    const float* x     = (const float*)d_in[0];   // [T,H]
    const float* Wg    = (const float*)d_in[1];   // [H,E]
    const float* W2    = (const float*)d_in[2];   // [H,D]
    const float* b2    = (const float*)d_in[3];   // [D]
    const float* A     = (const float*)d_in[4];   // [E,H,R]
    const float* Bm    = (const float*)d_in[5];   // [E,R,D]
    const float* scale = (const float*)d_in[6];   // [E]
    float* out = (float*)d_out;                   // [T,D] fp32

    char* ws = (char*)d_ws;
    size_t off = 0;
    unsigned short* xb     = (unsigned short*)(ws + off); off += (size_t)T_TOK * HDIM  * 2; // 64 MiB
    unsigned short* Wp     = (unsigned short*)(ws + off); off += (size_t)HDIM  * DDIM  * 2; //  8 MiB
    unsigned short* Apk    = (unsigned short*)(ws + off); off += (size_t)HDIM  * LRDIM * 2; //  4 MiB
    unsigned short* Bpk    = (unsigned short*)(ws + off); off += (size_t)LRDIM * DDIM  * 2; //  1 MiB
    unsigned short* hprime = (unsigned short*)(ws + off); off += (size_t)T_TOK * LRDIM * 2; //  8 MiB
    float*          cwsc   = (float*)(ws + off);          off += (size_t)T_TOK * EDIM  * 4; // .25 MiB

    router_kernel<<<T_TOK / 8, 256, 0, stream>>>(x, Wg, scale, cwsc);
    cvt_f32_bf16_v8<<<4096, 256, 0, stream>>>(x, xb, (long)T_TOK * HDIM / 8);
    pack_W2<<<2048, 256, 0, stream>>>(W2, Wp);
    pack_A <<<1024, 256, 0, stream>>>(A,  Apk);
    pack_Bm<<<512,  256, 0, stream>>>(Bm, Bpk);
    // h': (T/128) blocks x 8 expert quads
    lora_h_kernel<<<(T_TOK / 128) * (LRDIM / 64), 256, 0, stream>>>(xb, Apk, cwsc, hprime);
    // out: (T/128) blocks x 16 quads
    out_kernel<<<(T_TOK / 128) * (DDIM / 64), 256, 0, stream>>>(xb, Wp, hprime, Bpk, b2, out);
}
